// Decoder_71554155151955
// MI455X (gfx1250) — matrix operand order repaired and checked by reading, not compiled, not run on hardware
//
#include <hip/hip_runtime.h>
#include <math.h>

typedef __attribute__((ext_vector_type(16))) _Float16 v16h;
typedef __attribute__((ext_vector_type(8)))  _Float16 h8;
typedef __attribute__((ext_vector_type(8)))  float    v8f;

#define BRD   8   // breadth (switch branches)
#define WAVES 8   // waves per block (wave32 -> 256 threads)
#define MT    32  // M positions per wave (two 16x16 D tiles)

// ---------------------------------------------------------------------------
// Coupler: 1x1 conv -> softmax over BRD branches, per output pixel.
// relu/upsample applied to the *input read* (fused, no materialization).
// gate layout: [B][BRD][HO][WO]
// ---------------------------------------------------------------------------
__global__ __launch_bounds__(256) void gate_kernel(
    const float* __restrict__ src, const float* __restrict__ cw,
    const float* __restrict__ cb, float* __restrict__ gate,
    int Bn, int CI, int HI, int WI, int HO, int WO, int upsample, int relu)
{
  long tid = (long)blockIdx.x * blockDim.x + threadIdx.x;
  long total = (long)Bn * HO * WO;
  if (tid >= total) return;
  int ox = (int)(tid % WO);
  long t = tid / WO;
  int oy = (int)(t % HO);
  int b  = (int)(t / HO);
  int sy = upsample ? (oy >> 1) : oy;
  int sx = upsample ? (ox >> 1) : ox;
  const float* sp = src + (long)b * CI * HI * WI + (long)sy * WI + sx;

  float logit[BRD];
  #pragma unroll
  for (int v = 0; v < BRD; ++v) logit[v] = cb[v];
  for (int ci = 0; ci < CI; ++ci) {
    float xv = sp[(long)ci * HI * WI];
    if (relu) xv = fmaxf(xv, 0.0f);
    #pragma unroll
    for (int v = 0; v < BRD; ++v) logit[v] += cw[v * CI + ci] * xv;
  }
  float mx = logit[0];
  #pragma unroll
  for (int v = 1; v < BRD; ++v) mx = fmaxf(mx, logit[v]);
  float e[BRD]; float s = 0.0f;
  #pragma unroll
  for (int v = 0; v < BRD; ++v) { e[v] = __expf(logit[v] - mx); s += e[v]; }
  float inv = 1.0f / s;
  long gbase = ((long)b * BRD) * HO * WO + (long)oy * WO + ox;
  #pragma unroll
  for (int v = 0; v < BRD; ++v) gate[gbase + (long)v * HO * WO] = e[v] * inv;
}

// build a v16h A/B fragment from a 32-half LDS row:
// halves 0..7 <- row[kb .. kb+7], halves 8..15 <- row[kb2 .. kb2+7]
__device__ __forceinline__ v16h frag_from_row(const _Float16* row, int kb, int kb2)
{
  h8 lo = *(const h8*)(row + kb);
  h8 hi = *(const h8*)(row + kb2);
  return __builtin_shufflevector(lo, hi,
      0,1,2,3,4,5,6,7,8,9,10,11,12,13,14,15);
}

// ---------------------------------------------------------------------------
// SwitchedConv main conv as implicit GEMM on WMMA f32<-f16.
// Per wave: M = 32 positions (two 16-row D tiles), N = 16 output channels.
// Per K-step of 32: stage weights for all 8 branches (shared) + this wave's
// 32x32 im2col tile, then 16x v_wmma_f32_16x16x32_f16 (each B fragment reused
// by the two M-halves; independent accumulators allow co-execution).
// ---------------------------------------------------------------------------
__global__ __launch_bounds__(256) void sconv_wmma_kernel(
    const float* __restrict__ src,   // [B, CI, HI, WI] (pre-upsample dims)
    const float* __restrict__ Wt,    // [BRD, CO, CI, K, K]
    const float* __restrict__ bias,  // [BRD, CO]
    const float* __restrict__ gate,  // [B, BRD, HO, WO]
    float* __restrict__ out,         // [B, CO, HO, WO]
    int Bn, int CI, int HI, int WI,
    int CO, int HO, int WO,
    int K, int pad, int upsample, int relu, int accum)
{
  __shared__ alignas(16) _Float16 lds_w[BRD][16][32];    // [branch][n][k]  8KB
  __shared__ alignas(16) _Float16 lds_a[WAVES][MT][32];  // [wave][m][k]   16KB

  const int tid  = threadIdx.x;
  const int lane = tid & 31;
  const int wv   = tid >> 5;
  const int kk   = K * K;
  const int Ktot = CI * kk;
  const int Ksteps = (Ktot + 31) >> 5;
  const long Mtot = (long)Bn * HO * WO;
  const int c_base = blockIdx.y * 16;
  const long mbase = ((long)blockIdx.x * WAVES + wv) * MT;

  v8f acc[2][BRD];
  {
    v8f z = {};
    #pragma unroll
    for (int p = 0; p < 2; ++p)
      #pragma unroll
      for (int v = 0; v < BRD; ++v) acc[p][v] = z;
  }

  // A-gather assignment: lane owns one im2col row m = lane (all 32 k values)
  const long mg = mbase + lane;
  int ab = 0, aoy = 0, aox = 0;
  const bool mvalid = (mg < Mtot);
  if (mvalid) {
    aox = (int)(mg % WO);
    long t2 = mg / WO;
    aoy = (int)(t2 % HO);
    ab  = (int)(t2 / HO);
  }

  // Weight-stage assignment: tid -> branch v = tid>>5, n = (tid>>1)&15, k-half
  const int wvv = tid >> 5;
  const int wn  = (tid >> 1) & 15;
  const int wkh = (tid & 1) * 16;
  const int wc  = c_base + wn;
  const float* wp = (wc < CO) ? (Wt + ((long)wvv * CO + wc) * Ktot) : (const float*)0;

  for (int kc = 0; kc < Ksteps; ++kc) {
    // ---- stage weight chunk: lds_w[v][n][0..31] (f32 -> f16) ----
    {
      const int k0 = kc * 32 + wkh;
      #pragma unroll
      for (int j = 0; j < 16; ++j) {
        float w = 0.0f;
        const int k = k0 + j;
        if (wp && k < Ktot) w = wp[k];
        lds_w[wvv][wn][wkh + j] = (_Float16)w;
      }
    }
    // ---- stage im2col rows: lane writes its full 32-k row ----
    {
      const int k0 = kc * 32;
      int ci  = k0 / kk;
      int rem = k0 - ci * kk;
      int kh  = rem / K;
      int kw  = rem - kh * K;
      #pragma unroll
      for (int j = 0; j < 32; ++j) {
        float a = 0.0f;
        if (mvalid && ci < CI) {
          const int iy = aoy + kh - pad;   // coords in (possibly upsampled) input
          const int ix = aox + kw - pad;
          if (iy >= 0 && iy < HO && ix >= 0 && ix < WO) {
            const int sy = upsample ? (iy >> 1) : iy;  // fused nearest-2x
            const int sx = upsample ? (ix >> 1) : ix;
            a = src[(((long)ab * CI + ci) * HI + sy) * WI + sx];
            if (relu) a = fmaxf(a, 0.0f);
          }
        }
        lds_a[wv][lane][j] = (_Float16)a;  // note: per-lane row, j runs over k
        if (++kw == K) { kw = 0; if (++kh == K) { kh = 0; ++ci; } }
      }
    }
    __syncthreads();

    // ---- build fragments per documented 16-bit WMMA layouts, 16 WMMAs ----
    {
      const int mlo = lane & 15;
      const int kbA = (lane < 16) ? 0 : 8;   // A: h0-7 K=kb..kb+7, h8-15 K=16+kb..
      v16h a0 = frag_from_row(&lds_a[wv][mlo     ][0], kbA, 16 + kbA);
      v16h a1 = frag_from_row(&lds_a[wv][16 + mlo][0], kbA, 16 + kbA);

      const int kbB = (lane < 16) ? 0 : 8;   // B: h0-7 K=kb..kb+7, h8-15 K=16+kb..
      #pragma unroll
      for (int v = 0; v < BRD; ++v) {
        v16h bfrag = frag_from_row(&lds_w[v][lane & 15][0], kbB, 16 + kbB);
        acc[0][v] = __builtin_amdgcn_wmma_f32_16x16x32_f16(
            false, a0, false, bfrag, (short)0, acc[0][v], false, false);
        acc[1][v] = __builtin_amdgcn_wmma_f32_16x16x32_f16(
            false, a1, false, bfrag, (short)0, acc[1][v], false, false);
      }
    }
    __syncthreads();
  }

  // ---- epilogue: bias + gate-weighted branch sum (+ residual), store ----
  {
    const int c    = c_base + (lane & 15);
    const int moff = (lane < 16) ? 0 : 8;    // D: vgpr r -> M = r (+8 hi lanes)
    if (c < CO) {
      #pragma unroll
      for (int p = 0; p < 2; ++p) {
        #pragma unroll
        for (int r = 0; r < 8; ++r) {
          const long mg2 = mbase + p * 16 + moff + r;
          if (mg2 < Mtot) {
            const int ox = (int)(mg2 % WO);
            long t2 = mg2 / WO;
            const int oy = (int)(t2 % HO);
            const int b  = (int)(t2 / HO);
            float val = 0.0f;
            #pragma unroll
            for (int v = 0; v < BRD; ++v) {
              const float g = gate[(((long)b * BRD + v) * HO + oy) * WO + ox];
              val += g * (acc[p][v][r] + bias[v * CO + c]);
            }
            const long oidx = (((long)b * CO + c) * HO + oy) * WO + ox;
            if (accum) val += out[oidx];
            out[oidx] = val;
          }
        }
      }
    }
  }
}

// ---------------------------------------------------------------------------
static inline void launch_layer(const float* src, const float* Wt, const float* bias,
                                const float* cw, const float* cb,
                                float* gate, float* out,
                                int Bn, int CI, int HI, int WI,
                                int CO, int HO, int WO,
                                int K, int pad, int up, int relu, int accum,
                                hipStream_t s)
{
  const long total = (long)Bn * HO * WO;
  const int gblk = 256;
  const int ggrd = (int)((total + gblk - 1) / gblk);
  gate_kernel<<<ggrd, gblk, 0, s>>>(src, cw, cb, gate, Bn, CI, HI, WI, HO, WO, up, relu);

  const long mtiles = (total + MT - 1) / MT;
  dim3 grid((unsigned)((mtiles + WAVES - 1) / WAVES), (unsigned)((CO + 15) / 16));
  sconv_wmma_kernel<<<grid, 256, 0, s>>>(src, Wt, bias, gate, out,
                                         Bn, CI, HI, WI, CO, HO, WO,
                                         K, pad, up, relu, accum);
}

extern "C" void kernel_launch(void* const* d_in, const int* in_sizes, int n_in,
                              void* d_out, int out_size, void* d_ws, size_t ws_size,
                              hipStream_t stream)
{
  (void)in_sizes; (void)n_in; (void)out_size; (void)ws_size;
  const float* x = (const float*)d_in[0];
  auto P = [&](int i) { return (const float*)d_in[i]; };

  float* ws   = (float*)d_ws;
  float* h    = ws;                    // [8,128,64,64]   4,194,304
  float* m    = h  + 4194304;          // [8,64,64,64]    2,097,152
  float* u1   = m  + 2097152;          // [8,64,128,128]  8,388,608
  float* gate = u1 + 8388608;          // [8,8,256,256]   4,194,304 (max gate)

  // h = sconv(x, conv0, pad=1)
  launch_layer(x,  P(1),  P(2),  P(3),  P(4),  gate, h,
               8, 128, 64, 64, 128, 64, 64, 3, 1, 0, 0, 0, stream);
  // rb0: m = sconv(relu(h), c1, pad=1); h += sconv(relu(m), c2, pad=0)
  launch_layer(h,  P(5),  P(6),  P(7),  P(8),  gate, m,
               8, 128, 64, 64,  64, 64, 64, 3, 1, 0, 1, 0, stream);
  launch_layer(m,  P(9),  P(10), P(11), P(12), gate, h,
               8,  64, 64, 64, 128, 64, 64, 1, 0, 0, 1, 1, stream);
  // rb1
  launch_layer(h,  P(13), P(14), P(15), P(16), gate, m,
               8, 128, 64, 64,  64, 64, 64, 3, 1, 0, 1, 0, stream);
  launch_layer(m,  P(17), P(18), P(19), P(20), gate, h,
               8,  64, 64, 64, 128, 64, 64, 1, 0, 0, 1, 1, stream);
  // up1: u1 = sconv(up2x(relu(h)), pad=2)   (upsample fused into gather)
  launch_layer(h,  P(21), P(22), P(23), P(24), gate, u1,
               8, 128, 64, 64,  64, 128, 128, 5, 2, 1, 1, 0, stream);
  // up2: out = sconv(up2x(relu(u1)), pad=2)
  launch_layer(u1, P(25), P(26), P(27), P(28), gate, (float*)d_out,
               8,  64, 128, 128, 3, 256, 256, 5, 2, 1, 1, 0, stream);
}
